// TransformerEncoderLayer_6313601925203
// MI455X (gfx1250) — matrix operand order repaired and checked by reading, not compiled, not run on hardware
//
#include <hip/hip_runtime.h>

// ---------------------------------------------------------------------------
// MI455X (gfx1250, wave32) transformer encoder layer.
// All matmuls via v_wmma_f32_16x16x32_bf16 (bf16 in, f32 accumulate).
// GEMM staging: A tile via Tensor Data Mover (TENSOR_LOAD_TO_LDS, TENSORcnt),
// B tile via GLOBAL_LOAD_ASYNC_TO_LDS_B128 (ASYNCcnt); double-buffered LDS so
// both DMA engines overlap the WMMA stream.
// ---------------------------------------------------------------------------

typedef __attribute__((ext_vector_type(16))) __bf16 v16bf;
typedef __attribute__((ext_vector_type(8)))  float  v8f;
typedef __attribute__((ext_vector_type(4)))  unsigned int v4u;
typedef __attribute__((ext_vector_type(8)))  int v8i;
typedef __attribute__((ext_vector_type(4)))  int v4i;

union FragU { v16bf v; uint4 q[2]; };

__device__ __forceinline__ v8f vzero8() {
  v8f z = {0.f, 0.f, 0.f, 0.f, 0.f, 0.f, 0.f, 0.f};
  return z;
}

// round-to-nearest-even f32 -> bf16 (stored as ushort)
__device__ __forceinline__ unsigned short f2bfu(float f) {
  union { float f; unsigned int u; } c;
  c.f = f;
  unsigned int u = c.u;
  u += 0x7FFFu + ((u >> 16) & 1u);
  return (unsigned short)(u >> 16);
}

// Flat shared-memory pointer -> 32-bit LDS byte address (aperture truncation:
// low 32 bits of the flat address, CDNA5 ISA aperture table).
__device__ __forceinline__ unsigned ldsoff(const void* p) {
  return (unsigned)(uintptr_t)p;
}

// Async DMA: 16 bytes per lane, global -> LDS, tracked with ASYNCcnt.
__device__ __forceinline__ void async_ld_b128(unsigned lds, const void* g) {
  asm volatile("global_load_async_to_lds_b128 %0, %1, off"
               :: "v"(lds), "v"(g) : "memory");
}
__device__ __forceinline__ void wait_async0() {
  asm volatile("s_wait_asynccnt 0x0" ::: "memory");
}

// Tensor Data Mover: DMA a 2D tile (tile_d0 x tile_d1, 2-byte elements) from
// global (row stride stride_d0 elements) into LDS, packed linearly.
// D# layout per CDNA5 ISA sect. 8.3/8.4.
// 6-arg builtin form (clang-23 / therock-10.0 headers).
__device__ __forceinline__ void tdm_load_2d(unsigned lds_addr, const void* gaddr,
                                            unsigned tensor_d0,
                                            unsigned tensor_d1,
                                            unsigned tile_d0, unsigned tile_d1,
                                            unsigned stride_d0) {
  unsigned long long ga = (unsigned long long)(uintptr_t)gaddr;
  v4u g0;
  g0[0] = 1u;                                   // count=1, user descriptor
  g0[1] = lds_addr;                             // lds_addr [63:32]
  g0[2] = (unsigned)(ga & 0xFFFFFFFFu);         // global_addr[31:0]
  g0[3] = (unsigned)((ga >> 32) & 0x01FFFFFFu)  // global_addr[56:32]
          | (2u << 30);                         // type = 2 ("image")
  v8i g1;
  g1[0] = (int)(1u << 16);                      // workgroup_mask=0, data_size=1 (2B)
  g1[1] = (int)((tensor_d0 & 0xFFFFu) << 16);   // tensor_dim0[15:0] @ bit48
  g1[2] = (int)((tensor_d0 >> 16) | ((tensor_d1 & 0xFFFFu) << 16));
  g1[3] = (int)((tensor_d1 >> 16) | (tile_d0 << 16));  // tile_dim0 @ bit112
  g1[4] = (int)tile_d1;                         // tile_dim1 (tile_dim2 = 0)
  g1[5] = (int)stride_d0;                       // tensor_dim0_stride[31:0]
  g1[6] = 0;                                    // stride hi, dim1_stride lo
  g1[7] = 0;
  v4i z4 = {0, 0, 0, 0};                        // groups 2/3 unused (2D tensor)
  v8i z8 = {0, 0, 0, 0, 0, 0, 0, 0};
  __builtin_amdgcn_tensor_load_to_lds(g0, g1, z4, z4, z8, 0);
}
__device__ __forceinline__ void wait_tensor0() {
  __builtin_amdgcn_s_wait_tensorcnt(0);
}

// ---------------------------------------------------------------------------
// Packing kernels: f32 -> bf16, with layout rearrangement for WMMA B operands.
// ---------------------------------------------------------------------------

__global__ void pack_f32_bf16(const float* __restrict__ in,
                              unsigned short* __restrict__ out, int n) {
  int i = blockIdx.x * blockDim.x + threadIdx.x;
  if (i < n) out[i] = f2bfu(in[i]);
}

// wq [H, D, DH] -> Bp [(H*DH), D]  (col-major B: Bp[n*K + k], n = h*DH+e, k = d)
__global__ void pack_head_w(const float* __restrict__ w,
                            unsigned short* __restrict__ o, int Dm, int DHm,
                            int total) {
  int idx = blockIdx.x * blockDim.x + threadIdx.x;
  if (idx >= total) return;
  int n = idx / Dm, k = idx % Dm;
  int h = n / DHm, e = n % DHm;
  o[idx] = f2bfu(w[((size_t)h * Dm + k) * DHm + e]);
}

// W [K, Ncol] row-major -> Bp [Ncol, K]  (Bp[n*K + k] = W[k*Ncol + n])
__global__ void pack_transpose(const float* __restrict__ w,
                               unsigned short* __restrict__ o, int K, int Ncol,
                               int total) {
  int idx = blockIdx.x * blockDim.x + threadIdx.x;
  if (idx >= total) return;
  int n = idx / K, k = idx % K;
  o[idx] = f2bfu(w[(size_t)k * Ncol + n]);
}

// V [B*S, D] bf16 -> Vt [(b*H+h)*64 + e][S] bf16 (per-head transposed), so
// attention B-fragments for P.V are contiguous 16B loads.
__global__ void transpose_v_kernel(const unsigned short* __restrict__ Vg,
                                   unsigned short* __restrict__ Vt, int S,
                                   int Dm, int total) {
  int idx = blockIdx.x * blockDim.x + threadIdx.x;
  if (idx >= total) return;
  int s = idx & (S - 1);     // S = 2048 (pow2)
  int rest = idx / S;        // bh*64 + e
  int e = rest & 63;
  int bh = rest >> 6;
  int b = bh >> 4, h = bh & 15;  // H = 16
  Vt[idx] = Vg[(size_t)(b * S + s) * Dm + h * 64 + e];
}

// ---------------------------------------------------------------------------
// bf16 WMMA GEMM: C[M,N] = A[M,K] * B^T(stored [N,K]) + bias, optional ReLU.
// Block tile 256x128, K-step 32, 256 threads = 8 waves, each wave 64x64
// (4x4 WMMA accumulators).  LDS double-buffered: A tile streamed by the TDM
// (one descriptor per K-step, issued by wave 0), B tile by per-lane async DMA.
// ---------------------------------------------------------------------------

__global__ __launch_bounds__(256) void gemm_bf16_wmma(
    const unsigned short* __restrict__ A,   // [M,K] row-major bf16
    const unsigned short* __restrict__ Bp,  // [N,K] (column-major of C) bf16
    const float* __restrict__ bias,         // [N] or nullptr
    unsigned short* __restrict__ Cb,        // bf16 out or nullptr
    float* __restrict__ Cf,                 // f32 out or nullptr
    int M, int N, int K, int relu) {
  constexpr int BM = 256, BN = 128, BK = 32;
  __shared__ __align__(16) unsigned short As[2][BM * BK];  // 2 x 16KB
  __shared__ __align__(16) unsigned short Bs[2][BN * BK];  // 2 x 8KB

  const int tid = threadIdx.x;
  const int wave = tid >> 5, lane = tid & 31;
  const int hl = lane >> 4, l16 = lane & 15;
  const int wm = wave >> 1, wn = wave & 1;  // wave tile: rows wm*64, cols wn*64
  const size_t bm = (size_t)blockIdx.y * BM;
  const size_t bn = (size_t)blockIdx.x * BN;

  v8f acc[4][4];
#pragma unroll
  for (int mt = 0; mt < 4; mt++)
#pragma unroll
    for (int nt = 0; nt < 4; nt++) acc[mt][nt] = vzero8();

  // Stage one K-tile into LDS buffer `buf`: A via TDM, B via async DMA.
  auto stage = [&](int buf, int k0) {
    if (wave == 0) {  // wave-uniform branch: one TDM issue per tile
      tdm_load_2d(ldsoff(&As[buf][0]), &A[bm * K + k0],
                  (unsigned)(K - k0), (unsigned)M - (unsigned)bm,
                  (unsigned)BK, (unsigned)BM, (unsigned)K);
    }
#pragma unroll
    for (int i = 0; i < 2; i++) {  // B: 512 16B chunks / 256 threads
      int c = tid + i * 256;
      int row = c >> 2, seg = c & 3;
      async_ld_b128(ldsoff(&Bs[buf][row * BK + seg * 8]),
                    &Bp[(bn + row) * K + k0 + seg * 8]);
    }
  };

  stage(0, 0);
  wait_async0();
  wait_tensor0();
  __syncthreads();

  int buf = 0;
  for (int k0 = 0; k0 < K; k0 += BK) {
    if (k0 + BK < K) stage(buf ^ 1, k0 + BK);  // DMA next tile during compute

    // A fragments (16x32): lane l16 = row, elements hl*8..+7, hl*8+16..+23
    FragU a[4], bfr[4];
#pragma unroll
    for (int mt = 0; mt < 4; mt++) {
      const unsigned short* ap =
          &As[buf][(wm * 64 + mt * 16 + l16) * BK + hl * 8];
      a[mt].q[0] = *(const uint4*)(ap);
      a[mt].q[1] = *(const uint4*)(ap + 16);
    }
    // B fragments (32x16): lane l16 = col, K = hl*8..+7, hl*8+16..+23
#pragma unroll
    for (int nt = 0; nt < 4; nt++) {
      const unsigned short* bp =
          &Bs[buf][(wn * 64 + nt * 16 + l16) * BK + hl * 8];
      bfr[nt].q[0] = *(const uint4*)(bp);
      bfr[nt].q[1] = *(const uint4*)(bp + 16);
    }
#pragma unroll
    for (int mt = 0; mt < 4; mt++)
#pragma unroll
      for (int nt = 0; nt < 4; nt++)
        acc[mt][nt] = __builtin_amdgcn_wmma_f32_16x16x32_bf16(
            false, a[mt].v, false, bfr[nt].v, (short)0, acc[mt][nt], false,
            false);

    wait_async0();    // our async B chunks for next buffer landed
    wait_tensor0();   // wave 0's TDM for next buffer landed
    __syncthreads();  // everyone done reading buf / writing buf^1
    buf ^= 1;
  }

  // Epilogue. C layout: lane l holds M = hl*8 + r, N = l16.
#pragma unroll
  for (int mt = 0; mt < 4; mt++)
#pragma unroll
    for (int nt = 0; nt < 4; nt++)
#pragma unroll
      for (int r = 0; r < 8; r++) {
        size_t row = bm + wm * 64 + mt * 16 + hl * 8 + r;
        size_t col = bn + wn * 64 + nt * 16 + l16;
        float v = acc[mt][nt][r];
        if (bias) v += bias[col];
        if (relu) v = fmaxf(v, 0.f);
        if (Cf) Cf[row * N + col] = v;
        if (Cb) Cb[row * N + col] = f2bfu(v);
      }
}

// ---------------------------------------------------------------------------
// Flash attention: one block = 128 query rows of one (b,h); 4 waves, each wave
// owns 32 rows.  K fragments and (pre-transposed) V fragments stream straight
// from L2; P staged in per-wave LDS; no barriers needed in the main loop.
// ---------------------------------------------------------------------------

__global__ __launch_bounds__(128) void flash_attn_kernel(
    const unsigned short* __restrict__ Qg,  // [B*S, D] bf16, head h at cols h*64
    const unsigned short* __restrict__ Kg,  // [B*S, D] bf16
    const unsigned short* __restrict__ VT,  // [(b*H+h)*64 + e][S] bf16
    float* __restrict__ Og,                 // [B*S, D] f32
    int S, int Dm) {
  constexpr int KT = 64, DHk = 64;
  __shared__ __align__(16) unsigned short Pbuf[4][32 * KT];  // per-wave P

  const int b = blockIdx.z, h = blockIdx.y;
  const int tid = threadIdx.x, wave = tid >> 5, lane = tid & 31;
  const int hl = lane >> 4, l16 = lane & 15;
  const int base_tok = b * S;
  const int colh = h * DHk;
  const int wrow = blockIdx.x * 128 + wave * 32;
  const unsigned short* vbase = VT + (size_t)(b * 16 + h) * DHk * S;

  // Q fragments kept in registers: 2 m-tiles x 2 k-steps (dh=64 = 2*32)
  FragU qf[2][2];
#pragma unroll
  for (int mt = 0; mt < 2; mt++) {
    const unsigned short* qrow =
        Qg + (size_t)(base_tok + wrow + mt * 16 + l16) * Dm + colh;
#pragma unroll
    for (int ks = 0; ks < 2; ks++) {
      qf[mt][ks].q[0] = *(const uint4*)(qrow + ks * 32 + hl * 8);
      qf[mt][ks].q[1] = *(const uint4*)(qrow + ks * 32 + hl * 8 + 16);
    }
  }

  float m_run[16], l_run[16];
#pragma unroll
  for (int i = 0; i < 16; i++) {
    m_run[i] = -1e30f;
    l_run[i] = 0.f;
  }
  v8f oacc[2][4];
#pragma unroll
  for (int mt = 0; mt < 2; mt++)
#pragma unroll
    for (int nt = 0; nt < 4; nt++) oacc[mt][nt] = vzero8();

  for (int t0 = 0; t0 < S; t0 += KT) {
    if (t0 + KT < S) {  // hint next K tile toward the caches
      __builtin_prefetch(
          Kg + (size_t)(base_tok + t0 + KT + lane) * Dm + colh, 0, 2);
      __builtin_prefetch(vbase + (size_t)lane * S + t0 + KT, 0, 2);
    }

    // Scores: S = Q . K^T   (row-major K rows are exactly B-fragment columns)
    v8f sacc[2][4];
#pragma unroll
    for (int mt = 0; mt < 2; mt++)
#pragma unroll
      for (int nt = 0; nt < 4; nt++) sacc[mt][nt] = vzero8();

#pragma unroll
    for (int nt = 0; nt < 4; nt++) {
      const unsigned short* krow =
          Kg + (size_t)(base_tok + t0 + nt * 16 + l16) * Dm + colh;
#pragma unroll
      for (int ks = 0; ks < 2; ks++) {
        FragU kf;
        kf.q[0] = *(const uint4*)(krow + ks * 32 + hl * 8);
        kf.q[1] = *(const uint4*)(krow + ks * 32 + hl * 8 + 16);
#pragma unroll
        for (int mt = 0; mt < 2; mt++)
          sacc[mt][nt] = __builtin_amdgcn_wmma_f32_16x16x32_bf16(
              false, qf[mt][ks].v, false, kf.v, (short)0, sacc[mt][nt], false,
              false);
      }
    }

    // Online softmax update (scale 1/sqrt(64) = 0.125)
#pragma unroll
    for (int mt = 0; mt < 2; mt++) {
#pragma unroll
      for (int r = 0; r < 8; r++) {
        int i = mt * 8 + r;
        float tmax = -1e30f;
#pragma unroll
        for (int nt = 0; nt < 4; nt++) {
          float sv = sacc[mt][nt][r] * 0.125f;
          sacc[mt][nt][r] = sv;
          tmax = fmaxf(tmax, sv);
        }
#pragma unroll
        for (int o2 = 1; o2 < 16; o2 <<= 1)
          tmax = fmaxf(tmax, __shfl_xor(tmax, o2, 32));
        float mnew = fmaxf(m_run[i], tmax);
        float alpha = __expf(m_run[i] - mnew);
        float ps = 0.f;
#pragma unroll
        for (int nt = 0; nt < 4; nt++) {
          float p = __expf(sacc[mt][nt][r] - mnew);
          ps += p;
          // write P in A-fragment-friendly row-major layout [32 rows][64 keys]
          Pbuf[wave][(mt * 16 + hl * 8 + r) * KT + nt * 16 + l16] = f2bfu(p);
        }
#pragma unroll
        for (int o2 = 1; o2 < 16; o2 <<= 1) ps += __shfl_xor(ps, o2, 32);
        l_run[i] = l_run[i] * alpha + ps;
        m_run[i] = mnew;
#pragma unroll
        for (int nt = 0; nt < 4; nt++) oacc[mt][nt][r] *= alpha;
      }
    }

    // O += P . V   (A = P from per-wave LDS, B = V^T columns from global/L2)
#pragma unroll
    for (int ks = 0; ks < 2; ks++) {
      FragU pf[2];
#pragma unroll
      for (int mt = 0; mt < 2; mt++) {
        const unsigned short* prow =
            &Pbuf[wave][(mt * 16 + l16) * KT + ks * 32 + hl * 8];
        pf[mt].q[0] = *(const uint4*)(prow);
        pf[mt].q[1] = *(const uint4*)(prow + 16);
      }
#pragma unroll
      for (int nt = 0; nt < 4; nt++) {
        FragU vf;
        const unsigned short* vcol =
            vbase + (size_t)(nt * 16 + l16) * S + t0 + ks * 32 + hl * 8;
        vf.q[0] = *(const uint4*)(vcol);
        vf.q[1] = *(const uint4*)(vcol + 16);
#pragma unroll
        for (int mt = 0; mt < 2; mt++)
          oacc[mt][nt] = __builtin_amdgcn_wmma_f32_16x16x32_bf16(
              false, pf[mt].v, false, vf.v, (short)0, oacc[mt][nt], false,
              false);
      }
    }
  }

  // Normalize and write O (f32), heads concatenated on columns.
#pragma unroll
  for (int mt = 0; mt < 2; mt++)
#pragma unroll
    for (int nt = 0; nt < 4; nt++)
#pragma unroll
      for (int r = 0; r < 8; r++) {
        int i = mt * 8 + r;
        int row = base_tok + wrow + mt * 16 + hl * 8 + r;
        float v = oacc[mt][nt][r] / l_run[i];
        Og[(size_t)row * Dm + colh + nt * 16 + l16] = v;
      }
}

// ---------------------------------------------------------------------------
// res = x + LayerNorm(y); writes f32 (and optionally bf16 for next GEMM A).
// One block per token row (D = 1024, 256 threads x 4 elements).
// ---------------------------------------------------------------------------

__global__ __launch_bounds__(256) void ln_residual_kernel(
    const float* __restrict__ xres, const float* __restrict__ y,
    const float* __restrict__ gamma, const float* __restrict__ beta,
    float* __restrict__ outf, unsigned short* __restrict__ outb, int Dm) {
  const int row = blockIdx.x, tid = threadIdx.x;
  const int wave = tid >> 5, lane = tid & 31;
  const float* yr = y + (size_t)row * Dm;
  const float* xr = xres + (size_t)row * Dm;

  float v[4];
  float s = 0.f, s2 = 0.f;
#pragma unroll
  for (int i = 0; i < 4; i++) {
    v[i] = yr[tid + i * 256];
    s += v[i];
    s2 += v[i] * v[i];
  }
#pragma unroll
  for (int o2 = 16; o2 > 0; o2 >>= 1) {
    s += __shfl_xor(s, o2, 32);
    s2 += __shfl_xor(s2, o2, 32);
  }
  __shared__ float red[2][8];
  if (lane == 0) {
    red[0][wave] = s;
    red[1][wave] = s2;
  }
  __syncthreads();
  float ts = 0.f, ts2 = 0.f;
#pragma unroll
  for (int w = 0; w < 8; w++) {
    ts += red[0][w];
    ts2 += red[1][w];
  }
  float rD = 1.f / (float)Dm;
  float mu = ts * rD;
  float var = ts2 * rD - mu * mu;
  float inv = rsqrtf(var + 1e-5f);
#pragma unroll
  for (int i = 0; i < 4; i++) {
    int c = tid + i * 256;
    float ln = (v[i] - mu) * inv * gamma[c] + beta[c];
    float o = xr[c] + ln;
    outf[(size_t)row * Dm + c] = o;
    if (outb) outb[(size_t)row * Dm + c] = f2bfu(o);
  }
}

// ---------------------------------------------------------------------------
// Host-side orchestration.
// ---------------------------------------------------------------------------

extern "C" void kernel_launch(void* const* d_in, const int* in_sizes, int n_in,
                              void* d_out, int out_size, void* d_ws,
                              size_t ws_size, hipStream_t stream) {
  (void)in_sizes; (void)n_in; (void)out_size; (void)ws_size;

  constexpr int B = 4, S = 2048, D = 1024, H = 16, DHc = 64, DFF = 4096;
  constexpr int NT = B * S;  // 8192 tokens

  const float* x   = (const float*)d_in[0];
  const float* wq  = (const float*)d_in[1];
  const float* bq  = (const float*)d_in[2];
  const float* wk  = (const float*)d_in[3];
  const float* bk  = (const float*)d_in[4];
  const float* wv  = (const float*)d_in[5];
  const float* bv  = (const float*)d_in[6];
  const float* g1  = (const float*)d_in[7];
  const float* be1 = (const float*)d_in[8];
  const float* w1  = (const float*)d_in[9];
  const float* b1  = (const float*)d_in[10];
  const float* w2  = (const float*)d_in[11];
  const float* b2  = (const float*)d_in[12];
  const float* g2  = (const float*)d_in[13];
  const float* be2 = (const float*)d_in[14];
  float* out = (float*)d_out;

  char* wsp = (char*)d_ws;
  size_t off = 0;
  auto alloc = [&](size_t bytes) -> void* {
    void* p = wsp + off;
    off += (bytes + 255) & ~(size_t)255;
    return p;
  };

  unsigned short* xb    = (unsigned short*)alloc((size_t)NT * D * 2);
  unsigned short* Bq    = (unsigned short*)alloc((size_t)D * D * 2);
  unsigned short* Bk    = (unsigned short*)alloc((size_t)D * D * 2);
  unsigned short* Bv    = (unsigned short*)alloc((size_t)D * D * 2);
  unsigned short* B1    = (unsigned short*)alloc((size_t)D * DFF * 2);
  unsigned short* B2    = (unsigned short*)alloc((size_t)D * DFF * 2);
  unsigned short* Qg    = (unsigned short*)alloc((size_t)NT * D * 2);
  unsigned short* Kgb   = (unsigned short*)alloc((size_t)NT * D * 2);
  unsigned short* Vgb   = (unsigned short*)alloc((size_t)NT * D * 2);
  unsigned short* Vtb   = (unsigned short*)alloc((size_t)NT * D * 2);
  float*          attn  = (float*)alloc((size_t)NT * D * 4);
  float*          res1  = (float*)alloc((size_t)NT * D * 4);
  unsigned short* res1b = (unsigned short*)alloc((size_t)NT * D * 2);
  unsigned short* h1b   = (unsigned short*)alloc((size_t)NT * DFF * 2);
  float*          ff    = (float*)alloc((size_t)NT * D * 4);

  // 1) pack activations + weights to bf16 (weights in col-major B layout)
  {
    int nx = NT * D;
    pack_f32_bf16<<<(nx + 255) / 256, 256, 0, stream>>>(x, xb, nx);
    int nw = H * DHc * D;  // = D*D
    pack_head_w<<<(nw + 255) / 256, 256, 0, stream>>>(wq, Bq, D, DHc, nw);
    pack_head_w<<<(nw + 255) / 256, 256, 0, stream>>>(wk, Bk, D, DHc, nw);
    pack_head_w<<<(nw + 255) / 256, 256, 0, stream>>>(wv, Bv, D, DHc, nw);
    int nf = D * DFF;
    pack_transpose<<<(nf + 255) / 256, 256, 0, stream>>>(w1, B1, D, DFF, nf);
    pack_transpose<<<(nf + 255) / 256, 256, 0, stream>>>(w2, B2, DFF, D, nf);
  }

  // 2) Q/K/V projections: [NT,1024] = xb @ W + bias  (heads on columns)
  {
    dim3 grid(D / 128, NT / 256);
    gemm_bf16_wmma<<<grid, 256, 0, stream>>>(xb, Bq, bq, Qg, nullptr, NT, D, D, 0);
    gemm_bf16_wmma<<<grid, 256, 0, stream>>>(xb, Bk, bk, Kgb, nullptr, NT, D, D, 0);
    gemm_bf16_wmma<<<grid, 256, 0, stream>>>(xb, Bv, bv, Vgb, nullptr, NT, D, D, 0);
  }

  // 3) per-head transpose of V, then flash attention per (b, h, 128-row tile)
  {
    int nv = NT * D;
    transpose_v_kernel<<<(nv + 255) / 256, 256, 0, stream>>>(Vgb, Vtb, S, D, nv);
    dim3 grid(S / 128, H, B);
    flash_attn_kernel<<<grid, 128, 0, stream>>>(Qg, Kgb, Vtb, attn, S, D);
  }

  // 4) res1 = x + LN(attn)
  ln_residual_kernel<<<NT, 256, 0, stream>>>(x, attn, g1, be1, res1, res1b, D);

  // 5) FFN: h1 = relu(res1 @ W1 + b1); ff = h1 @ W2 + b2
  {
    dim3 grid1(DFF / 128, NT / 256);
    gemm_bf16_wmma<<<grid1, 256, 0, stream>>>(res1b, B1, b1, h1b, nullptr, NT,
                                              DFF, D, 1);
    dim3 grid2(D / 128, NT / 256);
    gemm_bf16_wmma<<<grid2, 256, 0, stream>>>(h1b, B2, b2, nullptr, ff, NT, D,
                                              DFF, 0);
  }

  // 6) out = res1 + LN(ff)
  ln_residual_kernel<<<NT, 256, 0, stream>>>(res1, ff, g2, be2, out, nullptr, D);
}
